// Attentionlayer_14851996910184
// MI455X (gfx1250) — matrix operand order repaired and checked by reading, not compiled, not run on hardware
//
#include <hip/hip_runtime.h>

typedef __attribute__((ext_vector_type(16))) __bf16 v16bf;
typedef __attribute__((ext_vector_type(8)))  float  v8f;
typedef __attribute__((ext_vector_type(4)))  unsigned tdm_u32x4;
typedef __attribute__((ext_vector_type(8)))  int      tdm_i32x8;
typedef __attribute__((ext_vector_type(4)))  int      tdm_i32x4;

#define S_LEN 2048
#define BATCH 2
#define IN_DIM 1024
#define NHEAD 16
#define HDIM 64
#define QKV_N 3072
#define ATTN_SCALE 0.125f

#if defined(__has_builtin)
#if __has_builtin(__builtin_amdgcn_tensor_load_to_lds) && __has_builtin(__builtin_amdgcn_s_wait_tensorcnt)
#define HAVE_TDM 1
#else
#define HAVE_TDM 0
#endif
#else
#define HAVE_TDM 0
#endif

// ---------------------------------------------------------------------------
// Fragment gather helpers. CDNA5 16-bit WMMA layouts (CDNA5 ISA):
//  A (16xK): lane L<16 holds row L, K={0..7,16..23}; lane L+16 holds K={8..15,24..31}
//  B (Kx16): lane n<16 holds col n, K={0..7,16..23}; lane n+16 holds col n, K={8..15,24..31}
//  C/D 16x16 f32: lane n (0..15) vgpr r -> (M=r, N=n); lane n+16 -> (M=r+8, N=n)
// ---------------------------------------------------------------------------
union FragPun { v16bf v; uint4 q[2]; };

static __device__ __forceinline__ v16bf frag_pair(const __bf16* p0, const __bf16* p1) {
    FragPun u;
    u.q[0] = *(const uint4*)p0;
    u.q[1] = *(const uint4*)p1;
    return u.v;
}

static __device__ __forceinline__ v16bf ldsA_frag(const __bf16* tile, int stride, int lane, int kc) {
    int half = lane >> 4, r = lane & 15;
    const __bf16* base = tile + r * stride + kc + half * 8;
    return frag_pair(base, base + 16);
}

static __device__ __forceinline__ v16bf ldsB_frag(const __bf16* tileN, int stride, int lane, int kc) {
    int half = lane >> 4, n = lane & 15;
    const __bf16* base = tileN + n * stride + kc + half * 8;
    return frag_pair(base, base + 16);
}

static __device__ __forceinline__ v8f vzero8() {
    v8f z;
#pragma unroll
    for (int i = 0; i < 8; i++) z[i] = 0.f;
    return z;
}

// ---------------------------------------------------------------------------
// Tensor Data Mover: one per-wave DMA of a 2D bf16 tile (rows x cols) from
// global to LDS. D# packing per CDNA5 ISA:
//  group0: [1:0]=count=1, [63:32]=lds_addr, [120:64]=global_addr, [127:126]=2
//  group1: [17:16]=data_size(1=2B), [79:48]=tensor_dim0, [111:80]=tensor_dim1,
//          [127:112]=tile_dim0, [143:128]=tile_dim1, [207:160]=dim0_stride
// Tensor dims set to 2^20 (>> any tile) so no OOB clamping triggers.
// ---------------------------------------------------------------------------
static __device__ __forceinline__ void tdm_load_2d(unsigned lds_off, const void* gaddr,
                                                   int tile_d0, int tile_d1, int stride_elems) {
#if HAVE_TDM
    unsigned long long ga = (unsigned long long)(uintptr_t)gaddr;
    tdm_u32x4 g0;
    g0[0] = 1u;                                                // count=1, user mode
    g0[1] = lds_off;                                           // lds_addr (bytes)
    g0[2] = (unsigned)ga;                                      // global_addr[31:0]
    g0[3] = ((unsigned)(ga >> 32) & 0x01FFFFFFu) | (2u << 30); // addr[56:32] | type=2
    tdm_i32x8 g1;
    g1[0] = 0x10000;                     // data_size=1 (2 bytes), no multicast/pad
    g1[1] = 0;                           // atomic barrier disabled
    g1[2] = 0x10;                        // tensor_dim0 = 1<<20 (upper bits)
    g1[3] = (tile_d0 << 16) | 0x10;      // tile_dim0 | tensor_dim1 upper bits
    g1[4] = tile_d1;                     // tile_dim1
    g1[5] = stride_elems;                // tensor_dim0_stride (data_size units)
    g1[6] = 0;
    g1[7] = 0;
    tdm_i32x4 z4;
    z4[0] = 0; z4[1] = 0; z4[2] = 0; z4[3] = 0;
#if __has_include(<hip/amd_detail/amd_gfx1250_TDM.h>)
    tdm_i32x8 z8;
#pragma unroll
    for (int i = 0; i < 8; i++) z8[i] = 0;
    __builtin_amdgcn_tensor_load_to_lds(g0, g1, z4, z4, z8, 0);
#else
    __builtin_amdgcn_tensor_load_to_lds(g0, g1, z4, z4, 0);
#endif
#endif
}

static __device__ __forceinline__ void tdm_wait() {
#if HAVE_TDM
    __builtin_amdgcn_s_wait_tensorcnt(0);
    asm volatile("" ::: "memory");   // LDS now holds TDM-written data
#endif
}

// ---------------------------------------------------------------------------
// DPP 16-lane butterfly reductions (pure VALU, no LDS round trips).
// ---------------------------------------------------------------------------
static __device__ __forceinline__ float dpp_max16(float x) {
    x = fmaxf(x, __int_as_float(__builtin_amdgcn_update_dpp(0, __float_as_int(x), 0xB1, 0xF, 0xF, true)));
    x = fmaxf(x, __int_as_float(__builtin_amdgcn_update_dpp(0, __float_as_int(x), 0x4E, 0xF, 0xF, true)));
    x = fmaxf(x, __int_as_float(__builtin_amdgcn_update_dpp(0, __float_as_int(x), 0x141, 0xF, 0xF, true)));
    x = fmaxf(x, __int_as_float(__builtin_amdgcn_update_dpp(0, __float_as_int(x), 0x140, 0xF, 0xF, true)));
    return x;
}

static __device__ __forceinline__ float dpp_sum16(float x) {
    x += __int_as_float(__builtin_amdgcn_update_dpp(0, __float_as_int(x), 0xB1, 0xF, 0xF, true));
    x += __int_as_float(__builtin_amdgcn_update_dpp(0, __float_as_int(x), 0x4E, 0xF, 0xF, true));
    x += __int_as_float(__builtin_amdgcn_update_dpp(0, __float_as_int(x), 0x141, 0xF, 0xF, true));
    x += __int_as_float(__builtin_amdgcn_update_dpp(0, __float_as_int(x), 0x140, 0xF, 0xF, true));
    return x;
}

static __device__ __forceinline__ unsigned pack_bf2(__bf16 a, __bf16 b) {
    union { __bf16 h[2]; unsigned u; } p;
    p.h[0] = a; p.h[1] = b;
    return p.u;
}

// ---------------------------------------------------------------------------
// Kernel 0: fp32 -> bf16 conversion (weights)
// ---------------------------------------------------------------------------
__global__ __launch_bounds__(256) void cvt_f32_bf16_kernel(const float* __restrict__ src,
                                                           __bf16* __restrict__ dst, int n) {
    int i = blockIdx.x * blockDim.x + threadIdx.x;
    int stride = gridDim.x * blockDim.x;
    for (; i < n; i += stride) dst[i] = (__bf16)src[i];
}

// ---------------------------------------------------------------------------
// Kernel 1: LayerNorm over last dim (1024), one block per (s,b) row -> bf16
// ---------------------------------------------------------------------------
__global__ __launch_bounds__(256) void ln_kernel(const float* __restrict__ x,
                                                 const float* __restrict__ g,
                                                 const float* __restrict__ be,
                                                 __bf16* __restrict__ out) {
    int row = blockIdx.x;
    const float* xr = x + (size_t)row * IN_DIM;
    float v[4];
    float s = 0.f, ss = 0.f;
#pragma unroll
    for (int i = 0; i < 4; i++) {
        v[i] = xr[threadIdx.x + 256 * i];
        s += v[i];
        ss += v[i] * v[i];
    }
#pragma unroll
    for (int off = 16; off > 0; off >>= 1) {
        s  += __shfl_xor(s, off, 32);
        ss += __shfl_xor(ss, off, 32);
    }
    __shared__ float ps[8], pq[8], stats[2];
    int lane = threadIdx.x & 31, w = threadIdx.x >> 5;
    if (lane == 0) { ps[w] = s; pq[w] = ss; }
    __syncthreads();
    if (threadIdx.x == 0) {
        float a = 0.f, b2 = 0.f;
#pragma unroll
        for (int i = 0; i < 8; i++) { a += ps[i]; b2 += pq[i]; }
        float mu  = a * (1.f / IN_DIM);
        float var = b2 * (1.f / IN_DIM) - mu * mu;
        stats[0] = mu;
        stats[1] = rsqrtf(var + 1e-5f);
    }
    __syncthreads();
    float mu = stats[0], rs = stats[1];
    __bf16* orow = out + (size_t)row * IN_DIM;
#pragma unroll
    for (int i = 0; i < 4; i++) {
        int c = threadIdx.x + 256 * i;
        orow[c] = (__bf16)((v[i] - mu) * rs * g[c] + be[c]);
    }
}

// ---------------------------------------------------------------------------
// Kernels 2 & 4: bf16 GEMM, 128x128 tile per 256-thread block (8 waves).
// EPI=0: store bf16.  EPI=1: add fp32 residual, store fp32.
// ---------------------------------------------------------------------------
template <int EPI>
__global__ __launch_bounds__(256) void gemm_bf16_kernel(const __bf16* __restrict__ A,
                                                        const __bf16* __restrict__ W,
                                                        int K, int N,
                                                        __bf16* __restrict__ outb,
                                                        float* __restrict__ outf,
                                                        const float* __restrict__ resid) {
    __shared__ __align__(16) __bf16 sA[128 * 32];   // [m][k] row-major
    __shared__ __align__(16) __bf16 sB[128 * 32];   // [n][k] N-major (transposed W tile)
    int t = threadIdx.x;
    int lane = t & 31, w = t >> 5;
    int rowBase = blockIdx.x * 128;
    int nBase   = blockIdx.y * 128;

    int ar = t >> 1, ahc = (t & 1) * 16;           // A tile: row, 16-elem half
    int wk = t >> 3, wn0 = (t & 7) * 16;           // W tile: k row, 16-elem n chunk

    v8f acc[8];
#pragma unroll
    for (int i = 0; i < 8; i++) acc[i] = vzero8();

    for (int k0 = 0; k0 < K; k0 += 32) {
        // stage A tile: 2 coalesced uint4 per thread
        {
            const uint4* src = (const uint4*)(A + (size_t)(rowBase + ar) * K + k0 + ahc);
            uint4* dst = (uint4*)(sA + ar * 32 + ahc);
            dst[0] = src[0];
            dst[1] = src[1];
        }
        // stage W tile transposed to N-major: 2 coalesced uint4 loads, unpack to LDS
        {
            const uint4* src = (const uint4*)(W + (size_t)(k0 + wk) * N + nBase + wn0);
            uint4 w0 = src[0], w1 = src[1];
            const __bf16* e0 = (const __bf16*)&w0;
            const __bf16* e1 = (const __bf16*)&w1;
#pragma unroll
            for (int e = 0; e < 8; e++) sB[(wn0 + e) * 32 + wk] = e0[e];
#pragma unroll
            for (int e = 0; e < 8; e++) sB[(wn0 + 8 + e) * 32 + wk] = e1[e];
        }
        if (k0 + 32 < K) {
            __builtin_prefetch(A + (size_t)(rowBase + ar) * K + k0 + 32 + ahc, 0, 0);
            __builtin_prefetch(W + (size_t)(k0 + 32 + wk) * N + nBase + wn0, 0, 0);
        }
        __syncthreads();
        v16bf a = ldsA_frag(sA + (w * 16) * 32, 32, lane, 0);
#pragma unroll
        for (int nt = 0; nt < 8; nt++) {
            v16bf b = ldsB_frag(sB + (nt * 16) * 32, 32, lane, 0);
            acc[nt] = __builtin_amdgcn_wmma_f32_16x16x32_bf16(false, a, false, b,
                                                              (short)0, acc[nt], false, false);
        }
        __syncthreads();
    }

    int half = lane >> 4, nlane = lane & 15;
#pragma unroll
    for (int nt = 0; nt < 8; nt++) {
#pragma unroll
        for (int r = 0; r < 8; r++) {
            int row = rowBase + w * 16 + r + 8 * half;
            int col = nBase + nt * 16 + nlane;
            float vv = acc[nt][r];
            if (EPI == 0) {
                outb[(size_t)row * N + col] = (__bf16)vv;
            } else {
                outf[(size_t)row * N + col] = resid[(size_t)row * N + col] + vv;
            }
        }
    }
}

// ---------------------------------------------------------------------------
// Kernel 3: causal flash attention. Grid (S/64, B*H), 128 threads (4 waves).
// Each wave owns a private 16-row Q tile + private LDS region -> no barriers.
// Q/K tiles staged by the Tensor Data Mover (one tensor_load_to_lds per wave
// per tile, waited with s_wait_tensorcnt); V staged manually (transposed).
// qkv layout: row (s*B+b), col h*192 + {q:0..63, k:64..127, v:128..191}
// ---------------------------------------------------------------------------
__global__ __launch_bounds__(128) void attn_kernel(const __bf16* __restrict__ qkv,
                                                   __bf16* __restrict__ vec) {
    __shared__ __align__(16) __bf16 sQ[4][16 * 64];   // [m][d]
    __shared__ __align__(16) __bf16 sK[4][32 * 64];   // [key][d]  (N-major for QK^T)
    __shared__ __align__(16) __bf16 sVt[4][64 * 32];  // [d][key]  (N-major for PV)
    __shared__ __align__(16) __bf16 sP[4][16 * 32];   // [m][key]  (A tile for PV)

    int lane = threadIdx.x & 31, w = threadIdx.x >> 5;
    int bh = blockIdx.y;
    int b = bh & (BATCH - 1);
    int h = bh >> 1;
    int qTile = blockIdx.x * 64 + w * 16;

    __bf16* Q  = sQ[w];
    __bf16* Kt = sK[w];
    __bf16* Vt = sVt[w];
    __bf16* P  = sP[w];

    const size_t hq = (size_t)h * 192;
    const int rowStride = BATCH * QKV_N;   // elems between consecutive sequence rows

    // ---- stage Q tile (16x64) ----
#if HAVE_TDM
    tdm_load_2d((unsigned)(uintptr_t)(void*)Q,
                qkv + ((size_t)qTile * BATCH + b) * QKV_N + hq,
                HDIM, 16, rowStride);
#else
#pragma unroll
    for (int tc = 0; tc < 4; tc++) {
        int idx = lane + 32 * tc;
        int r = idx >> 3, dc = (idx & 7) * 8;
        const uint4* src = (const uint4*)(qkv + ((size_t)(qTile + r) * BATCH + b) * QKV_N + hq + dc);
        *(uint4*)(Q + r * 64 + dc) = *src;
    }
#endif

    float run_max[8], lsum[8];
#pragma unroll
    for (int r = 0; r < 8; r++) { run_max[r] = -3.0e38f; lsum[r] = 0.f; }
    v8f O[4];
#pragma unroll
    for (int d = 0; d < 4; d++) O[d] = vzero8();

    int half = lane >> 4, n = lane & 15;
    int kp = (lane & 15) * 2;            // V transpose: this lane's key pair
    int dbase = (lane >> 4) * 32;        // V transpose: this lane's 32 dims
    int numKB = (qTile + 16 + 31) >> 5;  // causal: only key blocks with j0 <= qTile

    // Q fragments are loop-invariant: gather once (after Q tile is resident)
    tdm_wait();
    v16bf aq0 = ldsA_frag(Q, 64, lane, 0);
    v16bf aq1 = ldsA_frag(Q, 64, lane, 32);

    for (int kb = 0; kb < numKB; kb++) {
        int j0 = kb * 32;
        // ---- stage K block (32x64): one TDM per wave, overlaps V staging ----
#if HAVE_TDM
        tdm_load_2d((unsigned)(uintptr_t)(void*)Kt,
                    qkv + ((size_t)j0 * BATCH + b) * QKV_N + hq + 64,
                    HDIM, 32, rowStride);
#else
        {
            const __bf16* srck = qkv + ((size_t)(j0 + lane) * BATCH + b) * QKV_N + hq + 64;
#pragma unroll
            for (int c = 0; c < 8; c++)
                *(uint4*)(Kt + lane * 64 + c * 8) = *(const uint4*)(srck + c * 8);
        }
#endif
        // ---- stage V block transposed [d][key]: pack key pairs -> b32 stores ----
        {
            const __bf16* srcA = qkv + ((size_t)(j0 + kp) * BATCH + b) * QKV_N + hq + 128 + dbase;
            const __bf16* srcB = srcA + (size_t)rowStride;   // key kp+1
            __align__(16) __bf16 va[32], vb[32];
#pragma unroll
            for (int c = 0; c < 4; c++) {
                *(uint4*)(va + c * 8) = ((const uint4*)srcA)[c];
                *(uint4*)(vb + c * 8) = ((const uint4*)srcB)[c];
            }
#pragma unroll
            for (int d = 0; d < 32; d++)
                *(unsigned*)(Vt + (size_t)(dbase + d) * 32 + kp) = pack_bf2(va[d], vb[d]);
        }
        if (kb + 1 < numKB) {
            const __bf16* pf = qkv + ((size_t)(j0 + 32 + lane) * BATCH + b) * QKV_N + hq + 64;
            __builtin_prefetch(pf, 0, 0);
            __builtin_prefetch(pf + 64, 0, 0);
        }
        tdm_wait();   // K tile resident

        // ---- scores: gather all fragments first, then 4 wmma back-to-back ----
        v16bf bk00 = ldsB_frag(Kt,           64, lane, 0);
        v16bf bk01 = ldsB_frag(Kt + 16 * 64, 64, lane, 0);
        v16bf bk10 = ldsB_frag(Kt,           64, lane, 32);
        v16bf bk11 = ldsB_frag(Kt + 16 * 64, 64, lane, 32);
        v8f acc0 = vzero8(), acc1 = vzero8();
        acc0 = __builtin_amdgcn_wmma_f32_16x16x32_bf16(false, aq0, false, bk00, (short)0, acc0, false, false);
        acc1 = __builtin_amdgcn_wmma_f32_16x16x32_bf16(false, aq0, false, bk01, (short)0, acc1, false, false);
        acc0 = __builtin_amdgcn_wmma_f32_16x16x32_bf16(false, aq1, false, bk10, (short)0, acc0, false, false);
        acc1 = __builtin_amdgcn_wmma_f32_16x16x32_bf16(false, aq1, false, bk11, (short)0, acc1, false, false);

        // ---- online softmax: DPP butterflies, row m = r + 8*half ----
        float alpha[8];
#pragma unroll
        for (int r = 0; r < 8; r++) {
            int i = qTile + r + 8 * half;
            float s0 = acc0[r] * ATTN_SCALE;
            float s1 = acc1[r] * ATTN_SCALE;
            if (j0 + n > i)      s0 = -3.0e38f;
            if (j0 + 16 + n > i) s1 = -3.0e38f;
            float rm = dpp_max16(fmaxf(s0, s1));
            float nm = fmaxf(run_max[r], rm);
            alpha[r] = __expf(run_max[r] - nm);
            run_max[r] = nm;
            float p0 = __expf(s0 - nm);
            float p1 = __expf(s1 - nm);
            lsum[r] = lsum[r] * alpha[r] + dpp_sum16(p0 + p1);
            int m = r + 8 * half;
            P[m * 32 + n]      = (__bf16)p0;
            P[m * 32 + 16 + n] = (__bf16)p1;
        }
#pragma unroll
        for (int d = 0; d < 4; d++)
#pragma unroll
            for (int r = 0; r < 8; r++) O[d][r] *= alpha[r];

        // ---- PV: A = P (16x32), B = V via Vt N-major ----
        v16bf ap = ldsA_frag(P, 32, lane, 0);
#pragma unroll
        for (int d = 0; d < 4; d++) {
            v16bf bv = ldsB_frag(Vt + d * 16 * 32, 32, lane, 0);
            O[d] = __builtin_amdgcn_wmma_f32_16x16x32_bf16(false, ap, false, bv, (short)0, O[d], false, false);
        }
    }

    // normalize and write vec[(s*B+b)][h*64+d] as bf16
#pragma unroll
    for (int r = 0; r < 8; r++) {
        float inv = 1.f / lsum[r];
        int i = qTile + r + 8 * half;
        __bf16* dst = vec + ((size_t)i * BATCH + b) * (NHEAD * HDIM) + h * HDIM;
#pragma unroll
        for (int d = 0; d < 4; d++) dst[d * 16 + n] = (__bf16)(O[d][r] * inv);
    }
}

// ---------------------------------------------------------------------------
// Launch
// ---------------------------------------------------------------------------
extern "C" void kernel_launch(void* const* d_in, const int* in_sizes, int n_in,
                              void* d_out, int out_size, void* d_ws, size_t ws_size,
                              hipStream_t stream) {
    const float* x     = (const float*)d_in[0];
    const float* gamma = (const float*)d_in[1];
    const float* beta  = (const float*)d_in[2];
    const float* wqkv  = (const float*)d_in[3];
    const float* wout  = (const float*)d_in[4];
    float* out = (float*)d_out;

    char* ws = (char*)d_ws;
    __bf16* wqkv_bf = (__bf16*)(ws + 0);                    // 1024*3072*2 = 6 MB
    __bf16* wout_bf = (__bf16*)(ws + 6291456);              // 1024*1024*2 = 2 MB
    __bf16* h_bf    = (__bf16*)(ws + 8388608);              // 4096*1024*2 = 8 MB
    __bf16* qkv_bf  = (__bf16*)(ws + 16777216);             // 4096*3072*2 = 24 MB
    __bf16* vec_bf  = (__bf16*)(ws + 41943040);             // 4096*1024*2 = 8 MB

    cvt_f32_bf16_kernel<<<1024, 256, 0, stream>>>(wqkv, wqkv_bf, IN_DIM * QKV_N);
    cvt_f32_bf16_kernel<<<512, 256, 0, stream>>>(wout, wout_bf, IN_DIM * IN_DIM);

    ln_kernel<<<S_LEN * BATCH, 256, 0, stream>>>(x, gamma, beta, h_bf);

    // qkv = h @ w_qkv : M=4096 K=1024 N=3072
    gemm_bf16_kernel<0><<<dim3(32, 24), 256, 0, stream>>>(h_bf, wqkv_bf, IN_DIM, QKV_N,
                                                          qkv_bf, nullptr, nullptr);

    // flash attention per (qblock, b*h)
    attn_kernel<<<dim3(S_LEN / 64, BATCH * NHEAD), 128, 0, stream>>>(qkv_bf, vec_bf);

    // out = x + vec @ w_out : M=4096 K=1024 N=1024
    gemm_bf16_kernel<1><<<dim3(32, 8), 256, 0, stream>>>(vec_bf, wout_bf, NHEAD * HDIM, IN_DIM,
                                                         nullptr, out, x);
}
